// BrdfNetwork_1408749273176
// MI455X (gfx1250) — matrix operand order repaired and checked by reading, not compiled, not run on hardware
//
#include <hip/hip_runtime.h>
#include <hip/hip_bf16.h>
#include <cstdint>
#include <cstddef>

// ---------------------------------------------------------------------------
// Types for CDNA5 WMMA (wave32).
// ---------------------------------------------------------------------------
typedef __attribute__((ext_vector_type(8)))  float  v8f;
typedef __attribute__((ext_vector_type(16))) __bf16 v16bf;
typedef __attribute__((ext_vector_type(8)))  __bf16 v8bf;

// Lambert SH coefficients per band l (repeated per m in the reference).
__device__ __constant__ float c_lamb[11] = {
  3.14159265358979f, 2.09439510239320f, 0.785398163397448f, 0.0f,
 -0.130899693899575f, 0.0f, 0.0490873852123405f, 0.0f,
 -0.0245436926061703f, 0.0f, 0.0143171540202660f };

__device__ __forceinline__ unsigned short f2bf(float f) {
  unsigned int u = __float_as_uint(f);
  u += 0x7FFFu + ((u >> 16) & 1u);       // round-to-nearest-even
  return (unsigned short)(u >> 16);
}

__device__ __forceinline__ float sigmoidf(float x) { return 1.f / (1.f + __expf(-x)); }

__device__ constexpr float dfactf(int x) { float r = 1.f; while (x > 1) { r *= (float)x; x -= 2; } return r; }
__device__ constexpr double csqrt_(double x) {
  double g = (x > 1.0) ? x : 1.0;
  for (int i = 0; i < 64; i++) g = 0.5 * (g + x / g);
  return g;
}
__device__ constexpr float Kcoef(int l, int m) {
  double fr = 1.0;
  for (int i = l - m + 1; i <= l + m; i++) fr *= (double)i;
  return (float)csqrt_(((2.0 * l + 1.0) / (4.0 * 3.14159265358979323846)) / fr);
}

// ---------------------------------------------------------------------------
// Kernel 1: SH basis (normals + reflection) x ref_SH, streamed once.
// d_out layout (floats): raw_rgb[3N] | albedo[3N] | diffuse[3N] | specular[3N]
//                        | light[3N] | irradiance[3N]
// ---------------------------------------------------------------------------
__global__ __launch_bounds__(256) void sh_light_kernel(
    const float* __restrict__ normals, const float* __restrict__ view,
    const float* __restrict__ feature, const float* __restrict__ refSH,
    float* __restrict__ out, int N)
{
  const int n = blockIdx.x * blockDim.x + threadIdx.x;
  if (n >= N) return;

  float nx = normals[n * 3 + 0], ny = normals[n * 3 + 1], nz = normals[n * 3 + 2];
  float vx = view[n * 3 + 0],    vy = view[n * 3 + 1],    vz = view[n * 3 + 2];
  float rn = rsqrtf(fmaxf(nx * nx + ny * ny + nz * nz, 1e-24f));
  nx *= rn; ny *= rn; nz *= rn;
  float rv = rsqrtf(fmaxf(vx * vx + vy * vy + vz * vz, 1e-24f));
  vx *= rv; vy *= rv; vz *= rv;

  const float dt = nx * vx + ny * vy + nz * vz;
  float wx = 2.f * nx * dt - vx, wy = 2.f * ny * dt - vy, wz = 2.f * nz * dt - vz;
  float rw = rsqrtf(fmaxf(wx * wx + wy * wy + wz * wz, 1e-24f));
  wx *= rw; wy *= rw; wz *= rw;

  const float f0 = feature[(size_t)n * 16];
  const float rough = (f0 > 20.f) ? f0 : log1pf(__expf(f0));

  float cN[11], sN[11], cW[11], sW[11];
  cN[0] = 1.f; sN[0] = 0.f; cW[0] = 1.f; sW[0] = 0.f;
#pragma unroll
  for (int m = 1; m <= 10; m++) {
    cN[m] = nx * cN[m - 1] - ny * sN[m - 1];
    sN[m] = nx * sN[m - 1] + ny * cN[m - 1];
    cW[m] = wx * cW[m - 1] - wy * sW[m - 1];
    sW[m] = wx * sW[m - 1] + wy * cW[m - 1];
  }
  float dec[11];
#pragma unroll
  for (int l = 0; l <= 10; l++) dec[l] = __expf(-0.5f * (float)(l * (l + 1)) * rough);

  float aI0 = 0.f, aI1 = 0.f, aI2 = 0.f, aL0 = 0.f, aL1 = 0.f, aL2 = 0.f;
  const float* __restrict__ ref = refSH + (size_t)n * 363;

#pragma unroll
  for (int m = 0; m <= 10; m++) {
    float QN2 = 0.f, QN1 = 0.f, QW2 = 0.f, QW1 = 0.f;
#pragma unroll
    for (int l = m; l <= 10; l++) {
      float QN, QW;
      if (l == m)            { QN = dfactf(2 * m - 1); QW = QN; }
      else if (l == m + 1)   { QN = (float)(2 * m + 1) * nz * QN1;
                               QW = (float)(2 * m + 1) * wz * QW1; }
      else {
        const float inv = 1.f / (float)(l - m);
        QN = ((float)(2 * l - 1) * nz * QN1 - (float)(l + m - 1) * QN2) * inv;
        QW = ((float)(2 * l - 1) * wz * QW1 - (float)(l + m - 1) * QW2) * inv;
      }
      const float fK  = (m == 0) ? Kcoef(l, 0) : 1.41421356237f * Kcoef(l, m);
      const float lam = c_lamb[l] * fK;
      const float dcy = dec[l] * fK;
      if (m == 0) {
        const int k = l * l + l;
        const float r0 = ref[k * 3 + 0], r1 = ref[k * 3 + 1], r2 = ref[k * 3 + 2];
        const float wI = QN * lam, wL = QW * dcy;
        aI0 = fmaf(wI, r0, aI0); aI1 = fmaf(wI, r1, aI1); aI2 = fmaf(wI, r2, aI2);
        aL0 = fmaf(wL, r0, aL0); aL1 = fmaf(wL, r1, aL1); aL2 = fmaf(wL, r2, aL2);
      } else {
        const int kp = l * l + l + m, km = l * l + l - m;
        const float p0 = ref[kp * 3 + 0], p1 = ref[kp * 3 + 1], p2 = ref[kp * 3 + 2];
        const float m0 = ref[km * 3 + 0], m1 = ref[km * 3 + 1], m2 = ref[km * 3 + 2];
        const float wIp = QN * cN[m] * lam, wIm = QN * sN[m] * lam;
        const float wLp = QW * cW[m] * dcy, wLm = QW * sW[m] * dcy;
        aI0 = fmaf(wIp, p0, fmaf(wIm, m0, aI0));
        aI1 = fmaf(wIp, p1, fmaf(wIm, m1, aI1));
        aI2 = fmaf(wIp, p2, fmaf(wIm, m2, aI2));
        aL0 = fmaf(wLp, p0, fmaf(wLm, m0, aL0));
        aL1 = fmaf(wLp, p1, fmaf(wLm, m1, aL1));
        aL2 = fmaf(wLp, p2, fmaf(wLm, m2, aL2));
      }
      QN2 = QN1; QN1 = QN; QW2 = QW1; QW1 = QW;
    }
  }

  const float irr[3] = { fmaxf(aI0, 0.f), fmaxf(aI1, 0.f), fmaxf(aI2, 0.f) };
  const float lig[3] = { fmaxf(aL0, 0.f), fmaxf(aL1, 0.f), fmaxf(aL2, 0.f) };
  const size_t NN = (size_t)N;
#pragma unroll
  for (int c = 0; c < 3; c++) {
    const float al = sigmoidf(feature[(size_t)n * 16 + 10 + c]);
    out[3 * NN  + (size_t)n * 3 + c] = al;            // albedo
    out[6 * NN  + (size_t)n * 3 + c] = al * irr[c];   // diffuse
    out[12 * NN + (size_t)n * 3 + c] = lig[c];        // sh_light (light half)
    out[15 * NN + (size_t)n * 3 + c] = irr[c];        // sh_light (irradiance half)
  }
}

// ---------------------------------------------------------------------------
// Kernel 2: weight conversion f32 -> bf16, WMMA B-fragment layout.
// frag[((nt*nK + kt)*32 + lane)*16 + j] = W[kt*32 + (lane<16?0:8) + (j&7) + (j<8?0:16)][nt*16 + lane%16]
// ---------------------------------------------------------------------------
__global__ __launch_bounds__(256) void prep_w(
    const float* __restrict__ W, unsigned short* __restrict__ frag,
    int K, int srcK, int srcN, int Nout)
{
  const int e = blockIdx.x * 256 + threadIdx.x;
  if (e >= K * Nout) return;
  const int j    = e & 15;
  const int lane = (e >> 4) & 31;
  const int rest = e >> 9;
  const int nK   = K >> 5;
  const int kt   = rest % nK;
  const int nt   = rest / nK;
  const int k = kt * 32 + ((lane < 16) ? 0 : 8) + (j & 7) + ((j < 8) ? 0 : 16);
  const int n = nt * 16 + (lane & 15);
  const float v = (k < srcK && n < srcN) ? W[(size_t)k * srcN + n] : 0.f;
  frag[e] = f2bf(v);
}

// ---------------------------------------------------------------------------
// WMMA layer: 64 rows (4 M-tiles) x NOUT cols. Each wave loads one B-fragment
// and reuses it across 4 M-tile accumulators. LDS activation rows are padded
// (SIN/SOUT strides) so the 16 rows read by one ds_load_b128 wave hit all 64
// banks conflict-free. K-loop unroll kept at 2 to avoid VGPR spills.
// ---------------------------------------------------------------------------
template <int KIN, int SIN, int NOUT, int SOUT, bool RELU, bool FINAL>
__device__ __forceinline__ void do_layer64(
    const unsigned short* __restrict__ frag, const float* __restrict__ bias, int biasN,
    const unsigned short* __restrict__ aIn, unsigned short* __restrict__ aOut,
    float* __restrict__ csOut, int wave, int lane)
{
  constexpr int nK     = KIN >> 5;
  constexpr int nTiles = NOUT >> 4;
  const int laneM = lane & 15;
  const int koff  = (lane < 16) ? 0 : 8;
  const int rlo   = (lane < 16) ? 0 : 8;

  if constexpr (nTiles >= 8) {
    constexpr int NT_PER_WAVE = nTiles / 8;
#pragma clang loop unroll(disable)
    for (int i = 0; i < NT_PER_WAVE; ++i) {
      const int nt = wave * NT_PER_WAVE + i;
      v8f acc[4];
#pragma unroll
      for (int mt = 0; mt < 4; ++mt) acc[mt] = (v8f){0.f,0.f,0.f,0.f,0.f,0.f,0.f,0.f};

      const unsigned short* __restrict__ fBase =
          frag + ((size_t)(nt * nK) * 32 + lane) * 16;
#pragma unroll 2
      for (int kt = 0; kt < nK; ++kt) {
        const v16bf B = *(const v16bf*)(fBase + (size_t)kt * 512);
        if (kt + 1 < nK)
          __builtin_prefetch((const void*)(fBase + (size_t)(kt + 1) * 512), 0, 1);
#pragma unroll
        for (int mt = 0; mt < 4; ++mt) {
          const unsigned short* pa =
              aIn + (size_t)(mt * 16 + laneM) * SIN + kt * 32 + koff;
          const v8bf lo = *(const v8bf*)(pa);
          const v8bf hi = *(const v8bf*)(pa + 16);
          const v16bf A = __builtin_shufflevector(lo, hi, 0, 1, 2, 3, 4, 5, 6, 7,
                                                          8, 9, 10, 11, 12, 13, 14, 15);
          acc[mt] = __builtin_amdgcn_wmma_f32_16x16x32_bf16(false, A, false, B,
                                                            (short)0, acc[mt], false, false);
        }
      }
      const int n = (nt << 4) + laneM;
      const float bv = (n < biasN) ? bias[n] : 0.f;
#pragma unroll
      for (int mt = 0; mt < 4; ++mt) {
        const int rbase = mt * 16 + rlo;
#pragma unroll
        for (int v = 0; v < 8; v++) {
          float x = acc[mt][v] + bv;
          if (RELU) x = fmaxf(x, 0.f);
          aOut[(size_t)(rbase + v) * SOUT + n] = f2bf(x);
        }
      }
    }
  } else {
    // Final 512->32 layer: 8 tiles (4 M x 2 N), exactly one per wave.
    const int mt = wave & 3;
    const int nt = wave >> 2;
    v8f acc = { 0.f, 0.f, 0.f, 0.f, 0.f, 0.f, 0.f, 0.f };
    const unsigned short* __restrict__ fBase =
        frag + ((size_t)(nt * nK) * 32 + lane) * 16;
    const unsigned short* __restrict__ aRow =
        aIn + (size_t)(mt * 16 + laneM) * SIN + koff;
#pragma unroll 4
    for (int kt = 0; kt < nK; ++kt) {
      const v16bf B = *(const v16bf*)(fBase + (size_t)kt * 512);
      const unsigned short* pa = aRow + kt * 32;
      const v8bf lo = *(const v8bf*)(pa);
      const v8bf hi = *(const v8bf*)(pa + 16);
      const v16bf A = __builtin_shufflevector(lo, hi, 0, 1, 2, 3, 4, 5, 6, 7,
                                                      8, 9, 10, 11, 12, 13, 14, 15);
      acc = __builtin_amdgcn_wmma_f32_16x16x32_bf16(false, A, false, B,
                                                    (short)0, acc, false, false);
    }
    const int n = (nt << 4) + laneM;
    const float bv = (n < biasN) ? bias[n] : 0.f;
    const int rbase = mt * 16 + rlo;
#pragma unroll
    for (int v = 0; v < 8; v++) {
      float x = acc[v] + bv;
      if (RELU) x = fmaxf(x, 0.f);
      csOut[(size_t)(rbase + v) * SOUT + n] = x;
    }
  }
}

// ---------------------------------------------------------------------------
// Kernel 3: MLP (8->512->512->512->512->27) + gating tail. 64 rows per block.
// LDS strides padded for bank-conflict-free A-fragment loads:
//   S512 = 520 halfs (1040 B = 260 dwords; 260 % 64 = 4 -> lane r uses banks
//   4r..4r+3, all 64 banks covered), S32 = 40 halfs (80 B = 20 dwords).
// ---------------------------------------------------------------------------
__global__ __launch_bounds__(256) void mlp_kernel(
    const float* __restrict__ normals, const float* __restrict__ view,
    const float* __restrict__ feature,
    const float* __restrict__ b0, const float* __restrict__ b1,
    const float* __restrict__ b2, const float* __restrict__ b3,
    const float* __restrict__ bc,
    const unsigned short* __restrict__ w0f, const unsigned short* __restrict__ w1f,
    const unsigned short* __restrict__ w2f, const unsigned short* __restrict__ w3f,
    const unsigned short* __restrict__ wcf,
    float* __restrict__ out, int N)
{
  constexpr int S512 = 520;   // padded row stride (halfs) for 512-wide activations
  constexpr int S32  = 40;    // padded row stride (halfs) for 32-wide input
  __shared__ unsigned short sA[64 * S512];   // 65 KB
  __shared__ unsigned short sB[64 * S512];   // 65 KB  (130 KB of 320 KB WGP LDS)
  const int tid  = threadIdx.x;
  const int wave = tid >> 5;
  const int lane = tid & 31;
  const int rowBase = blockIdx.x * 64;

  // Stage padded input h = [dot, view, normal, roughness] (K padded 8 -> 32)
  // into the first 64 x S32 bf16 region of sB.
  if (tid < 64) {
    const int g = rowBase + tid;
    float h[8] = { 0.f, 0.f, 0.f, 0.f, 0.f, 0.f, 0.f, 0.f };
    if (g < N) {
      const float nx = normals[g * 3 + 0], ny = normals[g * 3 + 1], nz = normals[g * 3 + 2];
      const float vx = view[g * 3 + 0],    vy = view[g * 3 + 1],    vz = view[g * 3 + 2];
      const float dt = nx * vx + ny * vy + nz * vz;
      const float f0 = feature[(size_t)g * 16];
      const float rough = (f0 > 20.f) ? f0 : log1pf(__expf(f0));
      h[0] = dt; h[1] = vx; h[2] = vy; h[3] = vz;
      h[4] = nx; h[5] = ny; h[6] = nz; h[7] = rough;
    }
#pragma unroll
    for (int c = 0; c < 8; c++) sB[tid * S32 + c] = f2bf(h[c]);
#pragma unroll
    for (int c = 8; c < S32; c++) sB[tid * S32 + c] = 0;
  }
  __syncthreads();
  do_layer64<32,  S32,  512, S512, true, false>(w0f, b0, 512, sB, sA, nullptr, wave, lane);
  __syncthreads();
  do_layer64<512, S512, 512, S512, true, false>(w1f, b1, 512, sA, sB, nullptr, wave, lane);
  __syncthreads();
  do_layer64<512, S512, 512, S512, true, false>(w2f, b2, 512, sB, sA, nullptr, wave, lane);
  __syncthreads();
  do_layer64<512, S512, 512, S512, true, false>(w3f, b3, 512, sA, sB, nullptr, wave, lane);
  __syncthreads();
  float* sCS = reinterpret_cast<float*>(sA);   // sA is dead here; reuse as f32 cs
  do_layer64<512, S512, 32, 32, false, true>(wcf, bc, 27, sB, nullptr, sCS, wave, lane);
  __syncthreads();

  // Tail: gates, specular, raw_rgb.
  if (tid < 64) {
    const int g = rowBase + tid;
    if (g < N) {
      const float* f = feature + (size_t)g * 16;
      const size_t NN = (size_t)N;
      float coe[9];
#pragma unroll
      for (int i = 0; i < 9; i++) coe[i] = f[1 + i];
      const float* cs = sCS + tid * 32;
      float rr = 0.f, gg = 0.f, bb = 0.f;
#pragma unroll
      for (int i = 0; i < 9; i++) {
        rr = fmaf(cs[i], coe[i], rr);
        gg = fmaf(cs[9 + i], coe[i], gg);
        bb = fmaf(cs[18 + i], coe[i], bb);
      }
      const float gate[3] = { sigmoidf(rr), sigmoidf(gg), sigmoidf(bb) };
#pragma unroll
      for (int c = 0; c < 3; c++) {
        const float sp = sigmoidf(f[13 + c]);
        const float li = out[12 * NN + (size_t)g * 3 + c];
        const float di = out[6 * NN  + (size_t)g * 3 + c];
        const float spec = sp * li * gate[c];
        out[9 * NN + (size_t)g * 3 + c] = spec;      // specular
        out[(size_t)g * 3 + c]          = di + spec; // raw_rgb
      }
    }
  }
}

// ---------------------------------------------------------------------------
// Host launch. Inputs: 0 normals, 1 view, 2 feature, 3 ref_SH, 4 w0, 5 b0,
// 6 w1, 7 b1, 8 w2, 9 b2, 10 w3, 11 b3, 12 wc, 13 bc.
// ---------------------------------------------------------------------------
extern "C" void kernel_launch(void* const* d_in, const int* in_sizes, int n_in,
                              void* d_out, int out_size, void* d_ws, size_t ws_size,
                              hipStream_t stream) {
  (void)n_in; (void)out_size; (void)ws_size;
  const int N = in_sizes[0] / 3;

  const float* normals = (const float*)d_in[0];
  const float* view    = (const float*)d_in[1];
  const float* feature = (const float*)d_in[2];
  const float* refSH   = (const float*)d_in[3];
  const float* w0 = (const float*)d_in[4];
  const float* b0 = (const float*)d_in[5];
  const float* w1 = (const float*)d_in[6];
  const float* b1 = (const float*)d_in[7];
  const float* w2 = (const float*)d_in[8];
  const float* b2 = (const float*)d_in[9];
  const float* w3 = (const float*)d_in[10];
  const float* b3 = (const float*)d_in[11];
  const float* wc = (const float*)d_in[12];
  const float* bc = (const float*)d_in[13];
  float* out = (float*)d_out;

  unsigned short* ws  = (unsigned short*)d_ws;
  unsigned short* w0f = ws;                   // 32 x 512
  unsigned short* w1f = w0f + 32 * 512;       // 512 x 512
  unsigned short* w2f = w1f + 512 * 512;
  unsigned short* w3f = w2f + 512 * 512;
  unsigned short* wcf = w3f + 512 * 512;      // 512 x 32

  prep_w<<<(32 * 512 + 255) / 256, 256, 0, stream>>>(w0, w0f, 32, 8, 512, 512);
  prep_w<<<(512 * 512 + 255) / 256, 256, 0, stream>>>(w1, w1f, 512, 512, 512, 512);
  prep_w<<<(512 * 512 + 255) / 256, 256, 0, stream>>>(w2, w2f, 512, 512, 512, 512);
  prep_w<<<(512 * 512 + 255) / 256, 256, 0, stream>>>(w3, w3f, 512, 512, 512, 512);
  prep_w<<<(512 * 32 + 255) / 256, 256, 0, stream>>>(wc, wcf, 512, 512, 27, 32);

  sh_light_kernel<<<(N + 255) / 256, 256, 0, stream>>>(normals, view, feature, refSH, out, N);

  mlp_kernel<<<(N + 63) / 64, 256, 0, stream>>>(
      normals, view, feature, b0, b1, b2, b3, bc,
      w0f, w1f, w2f, w3f, wcf, out, N);
}
